// AGNNConv_62414464745579
// MI455X (gfx1250) — matrix operand order repaired and checked by reading, not compiled, not run on hardware
//
#include <hip/hip_runtime.h>
#include <hip/hip_bf16.h>

// ---------- types ----------
typedef __attribute__((ext_vector_type(16))) _Float16 v16h;
typedef __attribute__((ext_vector_type(8)))  _Float16 v8h;
typedef __attribute__((ext_vector_type(8)))  float    v8f;

#define FEAT_DIM 64
#define EPS 1e-12f

// Order-preserving float -> unsigned encoding (for atomicMax-based segment max)
__device__ __forceinline__ unsigned enc_f32(float f) {
    unsigned b = __float_as_uint(f);
    return (b & 0x80000000u) ? ~b : (b | 0x80000000u);
}
__device__ __forceinline__ float dec_f32(unsigned u) {
    unsigned b = (u & 0x80000000u) ? (u & 0x7FFFFFFFu) : ~u;
    return __uint_as_float(b);
}

// ---------------------------------------------------------------------------
// Kernel 1: per-node L2 normalize, store as f16. One wave (32 lanes) per node.
// ---------------------------------------------------------------------------
__global__ void k_normalize(const float* __restrict__ feat,
                            _Float16* __restrict__ nh, int n_nodes) {
    int wave = (int)((blockIdx.x * (unsigned)blockDim.x + threadIdx.x) >> 5);
    int lane = threadIdx.x & 31;
    if (wave >= n_nodes) return;
    const float* row = feat + (size_t)wave * FEAT_DIM;
    float a = row[lane];
    float b = row[lane + 32];
    float s = a * a + b * b;
    #pragma unroll
    for (int off = 16; off > 0; off >>= 1) s += __shfl_xor(s, off, 32);
    float rn = 1.0f / fmaxf(sqrtf(s), EPS);
    _Float16* o = nh + (size_t)wave * FEAT_DIM;
    o[lane]      = (_Float16)(a * rn);
    o[lane + 32] = (_Float16)(b * rn);
}

// ---------------------------------------------------------------------------
// Kernel 2: init out = 0, denom = 0, m_enc = 0 (below enc(-inf))
// ---------------------------------------------------------------------------
__global__ void k_init(float* __restrict__ out, unsigned* __restrict__ m_enc,
                       float* __restrict__ denom, int n_nodes) {
    int i = (int)(blockIdx.x * (unsigned)blockDim.x + threadIdx.x);
    int total = n_nodes * FEAT_DIM;
    if (i < total) out[i] = 0.0f;
    if (i < n_nodes) { m_enc[i] = 0u; denom[i] = 0.0f; }
}

// ---------------------------------------------------------------------------
// Kernel 3: SDDMM via WMMA. One wave handles a tile of 16 edges.
//  A fragment (16x32 f16): row M = src[tile+ (lane&15)]
//    lanes 0-15 : halves 0-7 <- K=kb+0..7,   halves 8-15 <- K=kb+16..23, kb=0
//    lanes 16-31: same with kb=8
//  B fragment (32x16 f16): col N = dst[tile+ (lane&15)]
//    lanes 0-15 : halves 0-7 <- K=cb+0..7, halves 8-15 <- K=cb+16..23, cb=0; lanes 16-31: cb=8
//  Diagonal of the 16x16 f32 accumulator = per-edge cosine.
// ---------------------------------------------------------------------------
__global__ void k_sddmm_wmma(const _Float16* __restrict__ nh,
                             const int* __restrict__ src,
                             const int* __restrict__ dst,
                             const float* __restrict__ beta,
                             float* __restrict__ e_out,
                             unsigned* __restrict__ m_enc,
                             int n_edges) {
    int wave = (int)((blockIdx.x * (unsigned)blockDim.x + threadIdx.x) >> 5);
    int lane = threadIdx.x & 31;
    int tile = wave * 16;
    if (tile >= n_edges) return;               // wave-uniform exit: EXEC stays full

    int j    = lane & 15;
    int eidx = tile + j;
    if (eidx >= n_edges) eidx = n_edges - 1;   // clamp tail (idempotent duplicates)

    int r = src[eidx];
    int c = dst[eidx];
    const _Float16* ra = nh + (size_t)r * FEAT_DIM;
    const _Float16* rb = nh + (size_t)c * FEAT_DIM;

    int kb = (lane >> 4) << 3;    // 0 or 8   (A K-offset)
    int cb = (lane >> 4) << 3;    // 0 or 8   (B K-offset)

    v8f acc = {};
    #pragma unroll
    for (int kc = 0; kc < 2; ++kc) {
        int base = kc * 32;
        v8h a0 = *(const v8h*)(ra + base + kb);        // K = kb..kb+7
        v8h a1 = *(const v8h*)(ra + base + kb + 16);   // K = kb+16..kb+23
        v8h b0 = *(const v8h*)(rb + base + cb);        // K = cb..cb+7
        v8h b1 = *(const v8h*)(rb + base + cb + 16);   // K = cb+16..cb+23
        v16h A, B;
        #pragma unroll
        for (int h = 0; h < 8; ++h) {
            A[h] = a0[h]; A[h + 8] = a1[h];
            B[h] = b0[h]; B[h + 8] = b1[h];
        }
        acc = __builtin_amdgcn_wmma_f32_16x16x32_f16(
            /*neg_a=*/false, A, /*neg_b=*/false, B,
            /*c_mod=*/(short)0, acc, /*reuse_a=*/false, /*reuse_b=*/false);
    }

    // Extract diagonal: edge v -> (VGPR v, lane v) for v<8,
    //                   edge v+8 -> (VGPR v, lane v+24)
    float ev = 0.0f;
    #pragma unroll
    for (int v = 0; v < 8; ++v) {
        float dv = acc[v];
        ev = (lane == v || lane == v + 24) ? dv : ev;
    }
    bool owner = (lane < 8) || (lane >= 24);
    if (owner) {
        float scale = beta[0];                 // TEMP == 1.0
        float val = ev * scale;
        e_out[eidx] = val;
        atomicMax(&m_enc[c], enc_f32(val));
    }
}

// ---------------------------------------------------------------------------
// Kernel 4: ex = exp(e - m[dst]); denom[dst] += ex. One thread per edge.
// ---------------------------------------------------------------------------
__global__ void k_softmax_num(float* __restrict__ e_arr,
                              const int* __restrict__ dst,
                              const unsigned* __restrict__ m_enc,
                              float* __restrict__ denom, int n_edges) {
    int i = (int)(blockIdx.x * (unsigned)blockDim.x + threadIdx.x);
    if (i >= n_edges) return;
    int c = dst[i];
    float m  = dec_f32(m_enc[c]);
    float ex = __expf(e_arr[i] - m);
    e_arr[i] = ex;                             // overwrite with numerator
    unsafeAtomicAdd(&denom[c], ex);
}

// ---------------------------------------------------------------------------
// Kernel 5: out[dst] += (ex/denom[dst]) * feat[src]. One wave per edge.
// ---------------------------------------------------------------------------
__global__ void k_aggregate(const float* __restrict__ feat,
                            const int* __restrict__ src,
                            const int* __restrict__ dst,
                            const float* __restrict__ e_arr,
                            const float* __restrict__ denom,
                            float* __restrict__ out, int n_edges) {
    int wave = (int)((blockIdx.x * (unsigned)blockDim.x + threadIdx.x) >> 5);
    int lane = threadIdx.x & 31;
    if (wave >= n_edges) return;
    int c = dst[wave];
    float attn = e_arr[wave] / fmaxf(denom[c], EPS);
    const float* fr = feat + (size_t)src[wave] * FEAT_DIM;
    float* orow = out + (size_t)c * FEAT_DIM;
    unsafeAtomicAdd(&orow[lane],      attn * fr[lane]);
    unsafeAtomicAdd(&orow[lane + 32], attn * fr[lane + 32]);
}

// ---------------------------------------------------------------------------
extern "C" void kernel_launch(void* const* d_in, const int* in_sizes, int n_in,
                              void* d_out, int out_size, void* d_ws, size_t ws_size,
                              hipStream_t stream) {
    const float* feat = (const float*)d_in[0];
    const float* beta = (const float*)d_in[1];
    const int*   src  = (const int*)d_in[2];
    const int*   dst  = (const int*)d_in[3];
    float* out = (float*)d_out;

    const int n_nodes = in_sizes[0] / FEAT_DIM;
    const int n_edges = in_sizes[2];

    // workspace partition
    char* ws = (char*)d_ws;
    size_t off = 0;
    _Float16* nh = (_Float16*)(ws + off);
    off += (size_t)n_nodes * FEAT_DIM * sizeof(_Float16);
    off = (off + 255) & ~(size_t)255;
    unsigned* m_enc = (unsigned*)(ws + off);
    off += (size_t)n_nodes * sizeof(unsigned);
    off = (off + 255) & ~(size_t)255;
    float* denom = (float*)(ws + off);
    off += (size_t)n_nodes * sizeof(float);
    off = (off + 255) & ~(size_t)255;
    float* e_arr = (float*)(ws + off);

    const int BLK = 256;                         // 8 waves per block
    // 1) normalize -> f16 (one wave per node)
    {
        int waves = n_nodes;
        int grid = (waves + (BLK / 32) - 1) / (BLK / 32);
        k_normalize<<<grid, BLK, 0, stream>>>(feat, nh, n_nodes);
    }
    // 2) init out / m / denom
    {
        int total = n_nodes * FEAT_DIM;
        int grid = (total + BLK - 1) / BLK;
        k_init<<<grid, BLK, 0, stream>>>(out, m_enc, denom, n_nodes);
    }
    // 3) SDDMM via WMMA (one wave per 16-edge tile) + segment max
    {
        int tiles = (n_edges + 15) / 16;
        int grid = (tiles + (BLK / 32) - 1) / (BLK / 32);
        k_sddmm_wmma<<<grid, BLK, 0, stream>>>(nh, src, dst, beta, e_arr, m_enc, n_edges);
    }
    // 4) softmax numerator + denominator
    {
        int grid = (n_edges + BLK - 1) / BLK;
        k_softmax_num<<<grid, BLK, 0, stream>>>(e_arr, dst, m_enc, denom, n_edges);
    }
    // 5) weighted scatter-sum aggregation (one wave per edge)
    {
        int grid = (n_edges + (BLK / 32) - 1) / (BLK / 32);
        k_aggregate<<<grid, BLK, 0, stream>>>(feat, src, dst, e_arr, denom, out, n_edges);
    }
}
